// BaseSpectrogram1D_25254407700745
// MI455X (gfx1250) — matrix operand order repaired and checked by reading, not compiled, not run on hardware
//
#include <hip/hip_runtime.h>
#include <math.h>

// Problem constants (from reference)
#define LX      131072
#define MFRAMES 511
#define NWIN    512
#define KH      257
#define STRIDE_ 256
#define BATCH   128

#define CTP      18           // padded number of 16-wide column tiles (288 cols)
#define COLS_PAD (CTP * 16)   // 288
#define CPAIRS   (CTP / 2)    // 9 column-tile pairs -> 9 waves per block
#define KCHUNKS  16           // 512 / 32
#define NROWS    (BATCH * MFRAMES)   // 65408
#define ROWPAIRS (NROWS / 32)        // 2044 (exact: 65408 = 32*2044)
#define FRAG_HALFS (CTP * KCHUNKS * 512)  // halfs per B fragment buffer (re or im)

#define ASTRIDE  520          // LDS row stride in halfs (1040B == 4 dwords mod 64 banks)

typedef __attribute__((ext_vector_type(16))) _Float16 v16h;
typedef __attribute__((ext_vector_type(8)))  _Float16 v8h;
typedef __attribute__((ext_vector_type(8)))  float    v8f;

// ---------------------------------------------------------------------------
// Pass 0: per-batch mean of x  (128 rows x 131072)
// ---------------------------------------------------------------------------
__global__ void spect_mean_kernel(const float* __restrict__ x, float* __restrict__ means) {
    __shared__ float s[256];
    const int b = blockIdx.x;
    const float* row = x + (size_t)b * LX;
    float acc = 0.f;
    for (int i = threadIdx.x; i < LX; i += 256) acc += row[i];
    s[threadIdx.x] = acc;
    __syncthreads();
    for (int off = 128; off > 0; off >>= 1) {
        if (threadIdx.x < off) s[threadIdx.x] += s[threadIdx.x + off];
        __syncthreads();
    }
    if (threadIdx.x == 0) means[b] = s[0] * (1.0f / LX);
}

// ---------------------------------------------------------------------------
// Pass 1: windowed DFT B-matrix fragments in this code's contiguous-k (same as A) WMMA B (32x16)
// lane layout: lane l -> col = l&15, K = (l>>4)*16 + j, 16 contiguous
// halfs per lane per (col-tile, k-chunk) fragment.
// idx = ((ct*16 + kc)*32 + lane)*16 + j
// ---------------------------------------------------------------------------
__global__ void spect_bfrag_kernel(const float* __restrict__ w,
                                   const float* __restrict__ fm,
                                   _Float16* __restrict__ fre, _Float16* __restrict__ fim) {
    const int idx = blockIdx.x * blockDim.x + threadIdx.x;
    if (idx >= FRAG_HALFS) return;
    const int j    = idx & 15;
    const int lane = (idx >> 4) & 31;
    const int kc   = (idx >> 9) & 15;
    const int ct   = idx >> 13;
    const int n    = kc * 32 + (lane >> 4) * 16 + j;
    const int col  = ct * 16 + (lane & 15);
    float re = 0.f, im = 0.f;
    if (col < KH) {
        const float wn = w[n];
        re = wn * fm[2 * ((size_t)n * KH + col)];
        im = wn * fm[2 * ((size_t)n * KH + col) + 1];
    }
    fre[idx] = (_Float16)re;
    fim[idx] = (_Float16)im;
}

// ---------------------------------------------------------------------------
// Pass 2: WMMA GEMM. One block = 9 waves = one 32-row tile x 288 cols.
// Block stages 32 mean-subtracted f16 frame rows in LDS (bank-padded),
// each wave owns a 2x2 of 16x16 WMMA tiles (32 rows x 32 cols, re+im).
// K-loop: 4 ds_load_b128 (A) + 8 global_load_b128 (B) -> 8 WMMAs.
// ---------------------------------------------------------------------------
__global__ __launch_bounds__(288) void spect_gemm_kernel(
        const float*    __restrict__ x,
        const float*    __restrict__ means,
        const _Float16* __restrict__ fre,
        const _Float16* __restrict__ fim,
        float*          __restrict__ out) {
    __shared__ _Float16 ash[32 * ASTRIDE];   // 33,280 B

    const int tid  = threadIdx.x;
    const int rp   = blockIdx.x;             // 32-row tile index

    // ---- Stage A: 32 rows x 512 samples, f32 -> (x - mean) -> f16 in LDS ----
    // 4096 float4 elements, 288 threads.
    for (int i = tid; i < 32 * 512 / 4; i += 288) {
        const int r  = i >> 7;               // 128 float4 per row
        const int c4 = (i & 127) * 4;
        const int gr = rp * 32 + r;
        const int b  = gr / MFRAMES;
        const int m  = gr % MFRAMES;
        const float4 v = *(const float4*)(x + (size_t)b * LX + (size_t)m * STRIDE_ + c4);
        const float mu = means[b];
        _Float16* dst = &ash[r * ASTRIDE + c4];
        dst[0] = (_Float16)(v.x - mu);
        dst[1] = (_Float16)(v.y - mu);
        dst[2] = (_Float16)(v.z - mu);
        dst[3] = (_Float16)(v.w - mu);
    }
    __syncthreads();

    // ---- Per-wave 2x2 WMMA tile ----
    const int wv   = tid >> 5;               // 0..8 -> col-tile pair
    const int lane = tid & 31;
    const int lgrp = lane >> 4;
    const int lcol = lane & 15;
    const int ct0  = wv * 2;
    const int ct1  = ct0 + 1;

    // A fragment LDS offsets (A layout: row = lane&15; K = lgrp*16 + 0..15 contiguous)
    const int abase0 = lcol * ASTRIDE        + lgrp * 16;   // rows 0..15
    const int abase1 = (16 + lcol) * ASTRIDE + lgrp * 16;   // rows 16..31

    // B fragment bases; k-chunk advances by 512 halfs
    const size_t bb0 = ((size_t)ct0 * KCHUNKS) * 512 + (size_t)lane * 16;
    const size_t bb1 = ((size_t)ct1 * KCHUNKS) * 512 + (size_t)lane * 16;

    v8f acc_re00 = {}, acc_im00 = {};
    v8f acc_re01 = {}, acc_im01 = {};
    v8f acc_re10 = {}, acc_im10 = {};
    v8f acc_re11 = {}, acc_im11 = {};

#pragma unroll 2
    for (int kc = 0; kc < KCHUNKS; ++kc) {
        const int ko = kc * 32;
        // A fragments from LDS: two aligned 16B runs per row
        const v8h a0lo = *(const v8h*)(&ash[abase0 + ko]);
        const v8h a0hi = *(const v8h*)(&ash[abase0 + ko + 8]);
        const v8h a1lo = *(const v8h*)(&ash[abase1 + ko]);
        const v8h a1hi = *(const v8h*)(&ash[abase1 + ko + 8]);
        const v16h a0 = __builtin_shufflevector(a0lo, a0hi,
                            0,1,2,3,4,5,6,7,8,9,10,11,12,13,14,15);
        const v16h a1 = __builtin_shufflevector(a1lo, a1hi,
                            0,1,2,3,4,5,6,7,8,9,10,11,12,13,14,15);

        const size_t kb = (size_t)kc * 512;
        const v16h br0 = *(const v16h*)(fre + bb0 + kb);
        const v16h bi0 = *(const v16h*)(fim + bb0 + kb);
        const v16h br1 = *(const v16h*)(fre + bb1 + kb);
        const v16h bi1 = *(const v16h*)(fim + bb1 + kb);

        acc_re00 = __builtin_amdgcn_wmma_f32_16x16x32_f16(false, a0, false, br0, (short)0, acc_re00, false, false);
        acc_im00 = __builtin_amdgcn_wmma_f32_16x16x32_f16(false, a0, false, bi0, (short)0, acc_im00, false, false);
        acc_re01 = __builtin_amdgcn_wmma_f32_16x16x32_f16(false, a0, false, br1, (short)0, acc_re01, false, false);
        acc_im01 = __builtin_amdgcn_wmma_f32_16x16x32_f16(false, a0, false, bi1, (short)0, acc_im01, false, false);
        acc_re10 = __builtin_amdgcn_wmma_f32_16x16x32_f16(false, a1, false, br0, (short)0, acc_re10, false, false);
        acc_im10 = __builtin_amdgcn_wmma_f32_16x16x32_f16(false, a1, false, bi0, (short)0, acc_im10, false, false);
        acc_re11 = __builtin_amdgcn_wmma_f32_16x16x32_f16(false, a1, false, br1, (short)0, acc_re11, false, false);
        acc_im11 = __builtin_amdgcn_wmma_f32_16x16x32_f16(false, a1, false, bi1, (short)0, acc_im11, false, false);
    }

    // ---- Epilogue: C/D layout -> VGPR v holds row (lgrp*8 + v), col = lane&15 ----
    const int col0 = ct0 * 16 + lcol;
    const int col1 = ct1 * 16 + lcol;

#pragma unroll
    for (int v = 0; v < 8; ++v) {
        const int r0 = rp * 32 + lgrp * 8 + v;       // row in first tile
        const int r1 = r0 + 16;                      // row in second tile
        if (col0 < KH) {
            const float re0 = acc_re00[v], im0 = acc_im00[v];
            out[(size_t)r0 * KH + col0] = sqrtf(re0 * re0 + im0 * im0);
            const float re1 = acc_re10[v], im1 = acc_im10[v];
            out[(size_t)r1 * KH + col0] = sqrtf(re1 * re1 + im1 * im1);
        }
        if (col1 < KH) {
            const float re0 = acc_re01[v], im0 = acc_im01[v];
            out[(size_t)r0 * KH + col1] = sqrtf(re0 * re0 + im0 * im0);
            const float re1 = acc_re11[v], im1 = acc_im11[v];
            out[(size_t)r1 * KH + col1] = sqrtf(re1 * re1 + im1 * im1);
        }
    }
}

// ---------------------------------------------------------------------------
extern "C" void kernel_launch(void* const* d_in, const int* in_sizes, int n_in,
                              void* d_out, int out_size, void* d_ws, size_t ws_size,
                              hipStream_t stream) {
    const float* x  = (const float*)d_in[0];   // [128, 131072] f32
    const float* w  = (const float*)d_in[1];   // [512] f32
    const float* fm = (const float*)d_in[2];   // [512, 257] complex64 (interleaved)
    float* out = (float*)d_out;                // [128, 511, 257] f32

    // Workspace layout
    float* means = (float*)d_ws;                       // 128 f32
    _Float16* fre = (_Float16*)((char*)d_ws + 4096);   // CTP*16*512 halfs
    _Float16* fim = fre + FRAG_HALFS;                  // CTP*16*512 halfs

    spect_mean_kernel<<<BATCH, 256, 0, stream>>>(x, means);
    spect_bfrag_kernel<<<(FRAG_HALFS + 255) / 256, 256, 0, stream>>>(w, fm, fre, fim);
    spect_gemm_kernel<<<ROWPAIRS, 288, 0, stream>>>(x, means, fre, fim, out);
}
